// Head_62191126446202
// MI455X (gfx1250) — matrix operand order repaired and checked by reading, not compiled, not run on hardware
//
#include <hip/hip_runtime.h>
#include <hip/hip_bf16.h>
#include <stdint.h>

// ---------------------------------------------------------------------------
// Single-head causal attention for MI455X (gfx1250, wave32, WMMA).
//   B=8, T=2048, C=1024, H=64
//   out = softmax(mask(q k^T * sqrt(C))) v,  q/k/v = x @ W{q,k,v}
// Phase 1 (proj): fused bf16 WMMA projection, x read once from HBM; packed
//   weights staged per-K-step into LDS by the Tensor Data Mover (TDM),
//   double-buffered (TENSORcnt).  An opaque never-taken store keeps the
//   compiler from folding reads of the DMA-written LDS (it can't see the
//   TDM's writes), while still lowering them to plain ds_load_b128.
// Phase 2 (attn): flash attention per wave, q/k/vT (6MB bf16) L2-resident;
//   P transpose through a wave-private LDS slice.
// ---------------------------------------------------------------------------

typedef __bf16 bf16_t;
typedef __attribute__((ext_vector_type(16))) __bf16 v16bf;
typedef __attribute__((ext_vector_type(8)))  __bf16 v8bf;
typedef __attribute__((ext_vector_type(8)))  float    v8f;
typedef unsigned int u32x4 __attribute__((ext_vector_type(4)));
typedef int          i32x4 __attribute__((ext_vector_type(4)));
typedef int          i32x8 __attribute__((ext_vector_type(8)));

#define NEG_BIG (-3.0e38f)

#if __has_builtin(__builtin_amdgcn_tensor_load_to_lds) && \
    __has_builtin(__builtin_amdgcn_s_wait_tensorcnt)
#define USE_TDM 1
#else
#define USE_TDM 0
#endif

__device__ __forceinline__ v16bf wmma_pack(v8bf lo, v8bf hi) {
  v16bf r;
#pragma unroll
  for (int i = 0; i < 8; ++i) { r[i] = lo[i]; r[i + 8] = hi[i]; }
  return r;
}

// A fragment (16x32, 16-bit): lane (m = lane&15, half = lane>>4) holds
// K = {half*16 + e} for e=0..15.
__device__ __forceinline__ v16bf load_a_bf16(const bf16_t* base, int ld, int row0, int k0) {
  const int lane = threadIdx.x & 31;
  const int hf = lane >> 4, m = lane & 15;
  const bf16_t* p = base + (size_t)(row0 + m) * ld + k0 + hf * 16;
  v8bf lo = *(const v8bf*)p;        // K = k0 + hf*16 + 0..7
  v8bf hi = *(const v8bf*)(p + 8);  // K = k0 + hf*16 + 8..15
  return wmma_pack(lo, hi);
}

// Same A fragment sourced from row-major f32 (convert to bf16 on the fly).
__device__ __forceinline__ v16bf load_a_f32(const float* base, int ld, int row0, int k0) {
  const int lane = threadIdx.x & 31;
  const int hf = lane >> 4, m = lane & 15;
  const float* p = base + (size_t)(row0 + m) * ld + k0 + hf * 16;
  const float4* p4 = (const float4*)p;
  float4 c0 = p4[0], c1 = p4[1];   // K offsets 0..7
  float4 c2 = p4[2], c3 = p4[3];   // K offsets 8..15
  v16bf r;
  r[0]  = (bf16_t)c0.x; r[1]  = (bf16_t)c0.y; r[2]  = (bf16_t)c0.z; r[3]  = (bf16_t)c0.w;
  r[4]  = (bf16_t)c1.x; r[5]  = (bf16_t)c1.y; r[6]  = (bf16_t)c1.z; r[7]  = (bf16_t)c1.w;
  r[8]  = (bf16_t)c2.x; r[9]  = (bf16_t)c2.y; r[10] = (bf16_t)c2.z; r[11] = (bf16_t)c2.w;
  r[12] = (bf16_t)c3.x; r[13] = (bf16_t)c3.y; r[14] = (bf16_t)c3.z; r[15] = (bf16_t)c3.w;
  return r;
}

// B fragment (32x16, 16-bit): lane n<16 holds column n, K=0..15; lanes 16..31
// hold K=16..31.  Caller passes this lane's 16 contiguous bf16 (K ascending).
__device__ __forceinline__ v16bf load_b_contig(const bf16_t* p) {
  v8bf lo = *(const v8bf*)p;
  v8bf hi = *(const v8bf*)(p + 8);
  return wmma_pack(lo, hi);
}

__device__ __forceinline__ v8f wmma_bf16(v16bf a, v16bf b, v8f c) {
  return __builtin_amdgcn_wmma_f32_16x16x32_bf16(false, a, false, b, (short)0, c, false, false);
}

#if USE_TDM
// 1-D TDM copy: ndw dwords from global gsrc -> LDS byte offset lds_off.
// Descriptor per CDNA5 ISA (data_size=4B).
__device__ __forceinline__ void tdm_load_1d(const void* gsrc, unsigned lds_off,
                                            unsigned ndw) {
  unsigned long long ga = (unsigned long long)(uintptr_t)gsrc;
  u32x4 g0;
  g0[0] = 1u;                                        // count=1 (valid, load)
  g0[1] = lds_off;                                   // lds_addr (bytes)
  g0[2] = (unsigned)(ga & 0xFFFFFFFFu);              // global_addr[31:0]
  g0[3] = (unsigned)((ga >> 32) & 0x1FFFFFFu) | (2u << 30);  // [56:32] | type=2
  i32x8 g1;
  g1[0] = (int)(2u << 16);                           // wg_mask=0, data_size=2 (4B)
  g1[1] = (int)((ndw & 0xFFFFu) << 16);              // tensor_dim0[15:0]
  g1[2] = (int)(((ndw >> 16) & 0xFFFFu) | (1u << 16)); // tensor_dim0[31:16], tensor_dim1=1
  g1[3] = (int)((ndw & 0xFFFFu) << 16);              // tile_dim0 = ndw
  g1[4] = 1;                                         // tile_dim1=1, tile_dim2 unused
  g1[5] = (int)ndw;                                  // tensor_dim0_stride[31:0]
  g1[6] = 0;                                         // stride0 hi, stride1 lo
  g1[7] = 0;
  i32x4 g2; g2[0] = 1; g2[1] = 1; g2[2] = 0; g2[3] = 0;  // dims 2/3 = 1
  i32x4 g3; g3[0] = 0; g3[1] = 0; g3[2] = 0; g3[3] = 0;
#if defined(__clang_major__) && __clang_major__ >= 23
  i32x8 g4 = {0, 0, 0, 0, 0, 0, 0, 0};
  __builtin_amdgcn_tensor_load_to_lds(g0, g1, g2, g3, g4, 0);
#else
  __builtin_amdgcn_tensor_load_to_lds(g0, g1, g2, g3, 0);
#endif
}
#endif  // USE_TDM

// ---------------------------------------------------------------------------
// Kernel 1: repack Wk/Wq/Wv [C,H] f32 into B-fragment-linear bf16, K-tile
// major so each K-step's tile (3 w * 4 nt * 512 = 12KB) is contiguous:
//   Wpack[kt][w][nt][lane][e] , e = K offset within 32-K tile for that lane.
// ---------------------------------------------------------------------------
__global__ void pack_w_kernel(const float* __restrict__ Wk,
                              const float* __restrict__ Wq,
                              const float* __restrict__ Wv,
                              bf16_t* __restrict__ Wpack, int C, int H) {
  const int KT = C / 32;
  const int total = KT * 3 * 4 * 32;                 // one thread per lane-slot
  int tid = blockIdx.x * blockDim.x + threadIdx.x;
  if (tid >= total) return;
  const int lane = tid & 31;
  const int nt   = (tid >> 5) & 3;
  const int w    = (tid >> 7) % 3;
  const int kt   = tid / 384;
  const float* W = (w == 0) ? Wk : ((w == 1) ? Wq : Wv);
  const int n = lane & 15, hf = lane >> 4;
  bf16_t* out = Wpack + (size_t)tid * 16;
#pragma unroll
  for (int e = 0; e < 16; ++e) {
    int k = kt * 32 + hf * 16 + e;
    out[e] = (bf16_t)W[(size_t)k * H + nt * 16 + n];
  }
}

// ---------------------------------------------------------------------------
// Kernel 2: fused projection. One wave = one 16-row tile; 8 waves/WG share a
// TDM-staged, double-buffered LDS copy of the per-K-step weight tile.
// Writes  kb,qb : [B,T,H] bf16 row-major;  vTb : [B,H,T] bf16 (transposed).
// ---------------------------------------------------------------------------
__global__ void proj_kernel(const float* __restrict__ x,
                            const bf16_t* __restrict__ Wpack,
                            bf16_t* __restrict__ kb,
                            bf16_t* __restrict__ qb,
                            bf16_t* __restrict__ vTb,
                            int Bn, int T, int C, int H) {
  __shared__ __align__(16) bf16_t wlds[2][6144];     // 2 x 12KB weight tiles
  const int tilesPerB = T / 16;
  const int wslot = threadIdx.x >> 5;
  const int wave = blockIdx.x * (blockDim.x >> 5) + wslot;
  const int b = wave / tilesPerB;   // uniform per block (8 waves, 8 | 128)
  if (b >= Bn) return;
  const int t0 = (wave % tilesPerB) * 16;
  const int lane = threadIdx.x & 31, hf = lane >> 4, n = lane & 15;
  const float* xb = x + (size_t)b * T * C;
  const int KT = C / 32;
  const unsigned TILE_BYTES = 3 * 4 * 512 * (unsigned)sizeof(bf16_t);  // 12288
  const unsigned TILE_DW = TILE_BYTES / 4;                             // 3072

  v8f acc[3][4];
#pragma unroll
  for (int w = 0; w < 3; ++w)
#pragma unroll
    for (int nt = 0; nt < 4; ++nt)
#pragma unroll
      for (int i = 0; i < 8; ++i) acc[w][nt][i] = 0.0f;

#if USE_TDM
  // The TDM writes wlds behind the compiler's back.  Plant a *potential*
  // store (opaque, never taken at runtime) so LLVM cannot prove wlds is
  // uninitialized and fold the ds loads to undef.  Reads stay plain
  // ds_load_b128 (no volatile / no flat SCOPE_SYS pessimization).
  {
    int poison = 0;
    asm volatile("" : "+v"(poison));           // compiler can't see it's 0
    if (poison) {
      wlds[0][(unsigned)poison & 6143u] = (bf16_t)1.0f;
      wlds[1][(unsigned)poison & 6143u] = (bf16_t)1.0f;
    }
  }
  // Prologue: TDM-stage tile 0 into buffer 0 (one DMA per workgroup).
  if (wslot == 0) {
    tdm_load_1d(Wpack, 0u, TILE_DW);
    __builtin_amdgcn_s_wait_tensorcnt(0);
  }
  __syncthreads();
  int buf = 0;
  for (int kt = 0; kt < KT; ++kt) {
    if (wslot == 0 && kt + 1 < KT)   // DMA next tile while we compute this one
      tdm_load_1d(Wpack + (size_t)(kt + 1) * 6144,
                  (unsigned)((buf ^ 1) * TILE_BYTES), TILE_DW);
    v16bf a = load_a_f32(xb, C, t0, kt * 32);
    const bf16_t* wb = wlds[buf] + (size_t)lane * 16;
#pragma unroll
    for (int w = 0; w < 3; ++w)
#pragma unroll
      for (int nt = 0; nt < 4; ++nt) {
        v16bf bfrag = load_b_contig(wb + (size_t)(w * 4 + nt) * 512);  // ds_load_b128
        acc[w][nt] = wmma_bf16(a, bfrag, acc[w][nt]);
      }
    if (wslot == 0) __builtin_amdgcn_s_wait_tensorcnt(0);
    __syncthreads();                 // next buffer ready; this buffer reusable
    buf ^= 1;
  }
#else
  // Fallback: cooperative dword staging, single buffer.
  for (int kt = 0; kt < KT; ++kt) {
    __syncthreads();
    {
      const uint32_t* src = (const uint32_t*)(Wpack + (size_t)kt * 6144);
      uint32_t* dst = (uint32_t*)&wlds[0][0];
      for (unsigned i = threadIdx.x; i < TILE_DW; i += blockDim.x) dst[i] = src[i];
    }
    __syncthreads();
    v16bf a = load_a_f32(xb, C, t0, kt * 32);
    const bf16_t* wb = wlds[0] + (size_t)lane * 16;
#pragma unroll
    for (int w = 0; w < 3; ++w)
#pragma unroll
      for (int nt = 0; nt < 4; ++nt) {
        v16bf bfrag = load_b_contig(wb + (size_t)(w * 4 + nt) * 512);
        acc[w][nt] = wmma_bf16(a, bfrag, acc[w][nt]);
      }
  }
#endif

  // C/D layout: VGPR r <-> row M = r + (lane>=16 ? 8 : 0), column N = lane&15.
#pragma unroll
  for (int nt = 0; nt < 4; ++nt) {
    const int h = nt * 16 + n;
#pragma unroll
    for (int r = 0; r < 8; ++r) {
      const int t = t0 + hf * 8 + r;
      const size_t idx = ((size_t)b * T + t) * H + h;
      kb[idx] = (bf16_t)acc[0][nt][r];
      qb[idx] = (bf16_t)acc[1][nt][r];
    }
    v8bf vv;
#pragma unroll
    for (int r = 0; r < 8; ++r) vv[r] = (bf16_t)acc[2][nt][r];
    *(v8bf*)(vTb + ((size_t)b * H + h) * T + t0 + hf * 8) = vv;  // contiguous in t
  }
}

// ---------------------------------------------------------------------------
// Kernel 3: flash attention. One wave = one 16-query tile; loops causal key
// blocks of 32.  S/P transposed C->A layout through a private LDS slice.
// (No workgroup barriers: waves have divergent causal trip counts.)
// ---------------------------------------------------------------------------
__global__ void attn_kernel(const bf16_t* __restrict__ qb,
                            const bf16_t* __restrict__ kb,
                            const bf16_t* __restrict__ vTb,
                            float* __restrict__ out,
                            int Bn, int T, int H, float scale) {
  __shared__ __align__(16) bf16_t ldsP[8 * 512];      // 8 waves * (16x32) bf16
  const int wslot = threadIdx.x >> 5;
  const int tilesPerB = T / 16;
  const int wave = blockIdx.x * (blockDim.x >> 5) + wslot;
  const int b = wave / tilesPerB;
  if (b >= Bn) return;
  const int t0 = (wave % tilesPerB) * 16;
  const int lane = threadIdx.x & 31, hf = lane >> 4, n = lane & 15;
  bf16_t* myP = ldsP + (size_t)wslot * 512;

  const bf16_t* qB = qb + (size_t)b * T * H;
  const bf16_t* kB = kb + (size_t)b * T * H;
  const bf16_t* vB = vTb + (size_t)b * H * T;

  // Q fragments (K = h of 0..31 and 32..63) live in registers for all blocks.
  const v16bf qf0 = load_a_bf16(qB, H, t0, 0);
  const v16bf qf1 = load_a_bf16(qB, H, t0, 32);

  v8f acc[4];
#pragma unroll
  for (int nt = 0; nt < 4; ++nt)
#pragma unroll
    for (int i = 0; i < 8; ++i) acc[nt][i] = 0.0f;
  float m_i[8], l_i[8];
#pragma unroll
  for (int r = 0; r < 8; ++r) { m_i[r] = NEG_BIG; l_i[r] = 0.0f; }

  const int sEnd = t0 + 16;                     // keys s < sEnd (causal)
  for (int s0 = 0; s0 < sEnd; s0 += 32) {
    if (s0 + 32 < sEnd) {                       // hint next key block into cache
      __builtin_prefetch(kB + (size_t)(s0 + 32 + n) * H, 0, 1);
      __builtin_prefetch(vB + (size_t)n * T + s0 + 32, 0, 1);
    }

    // ---- S = q k^T for two 16-key subtiles (K = H = 64 -> 2 WMMAs each) ----
    v8f S0, S1;
#pragma unroll
    for (int i = 0; i < 8; ++i) { S0[i] = 0.0f; S1[i] = 0.0f; }
    {
      v16bf kf;
      kf = load_b_contig(kB + (size_t)(s0 + n) * H + 0  + hf * 16);
      S0 = wmma_bf16(qf0, kf, S0);
      kf = load_b_contig(kB + (size_t)(s0 + n) * H + 32 + hf * 16);
      S0 = wmma_bf16(qf1, kf, S0);
      kf = load_b_contig(kB + (size_t)(s0 + 16 + n) * H + 0  + hf * 16);
      S1 = wmma_bf16(qf0, kf, S1);
      kf = load_b_contig(kB + (size_t)(s0 + 16 + n) * H + 32 + hf * 16);
      S1 = wmma_bf16(qf1, kf, S1);
    }

    // ---- scale + causal mask + online softmax (rows across 16-lane halves) -
#pragma unroll
    for (int r = 0; r < 8; ++r) {
      const int t = t0 + hf * 8 + r;
      float a0 = (s0 + n      <= t) ? S0[r] * scale : NEG_BIG;
      float a1 = (s0 + 16 + n <= t) ? S1[r] * scale : NEG_BIG;

      float mx = fmaxf(a0, a1);
      mx = fmaxf(mx, __shfl_xor(mx, 1, 32));
      mx = fmaxf(mx, __shfl_xor(mx, 2, 32));
      mx = fmaxf(mx, __shfl_xor(mx, 4, 32));
      mx = fmaxf(mx, __shfl_xor(mx, 8, 32));

      const float mn   = fmaxf(m_i[r], mx);
      const float corr = __expf(m_i[r] - mn);
      m_i[r] = mn;

      const float p0 = __expf(a0 - mn);
      const float p1 = __expf(a1 - mn);
      S0[r] = p0; S1[r] = p1;

      float rs = p0 + p1;
      rs += __shfl_xor(rs, 1, 32);
      rs += __shfl_xor(rs, 2, 32);
      rs += __shfl_xor(rs, 4, 32);
      rs += __shfl_xor(rs, 8, 32);
      l_i[r] = l_i[r] * corr + rs;

      acc[0][r] *= corr; acc[1][r] *= corr; acc[2][r] *= corr; acc[3][r] *= corr;
    }

    // ---- transpose P (C layout, f32) -> A layout (bf16) via private LDS ----
#pragma unroll
    for (int r = 0; r < 8; ++r) {
      const int m = hf * 8 + r;
      myP[m * 32 + n]      = (bf16_t)S0[r];
      myP[m * 32 + 16 + n] = (bf16_t)S1[r];
    }
    asm volatile("s_wait_dscnt 0x0" ::: "memory");   // wave-private LDS RAW
    v16bf pf;
    {
      const bf16_t* pr = myP + n * 32 + hf * 16;
      v8bf lo = *(const v8bf*)pr;         // local K = hf*16 + 0..7
      v8bf hi = *(const v8bf*)(pr + 8);   // local K = hf*16 + 8..15
      pf = wmma_pack(lo, hi);
    }

    // ---- O += P @ V  (V B-fragment contiguous thanks to [B,H,T] layout) ----
#pragma unroll
    for (int nt = 0; nt < 4; ++nt) {
      v16bf vf = load_b_contig(vB + (size_t)(nt * 16 + n) * T + s0 + hf * 16);
      acc[nt] = wmma_bf16(pf, vf, acc[nt]);
    }
  }

  // ---- epilogue: O / l, store f32 [B,T,H] ----
#pragma unroll
  for (int nt = 0; nt < 4; ++nt) {
    const int h = nt * 16 + n;
#pragma unroll
    for (int r = 0; r < 8; ++r) {
      const int t = t0 + hf * 8 + r;
      out[((size_t)b * T + t) * H + h] = acc[nt][r] / l_i[r];
    }
  }
}

// ---------------------------------------------------------------------------
extern "C" void kernel_launch(void* const* d_in, const int* in_sizes, int n_in,
                              void* d_out, int out_size, void* d_ws, size_t ws_size,
                              hipStream_t stream) {
  const int B = 8, T = 2048, C = 1024, H = 64;
  // setup_inputs() dict order: x, Wk, Wq, Wv
  const float* x  = (const float*)d_in[0];
  const float* Wk = (const float*)d_in[1];
  const float* Wq = (const float*)d_in[2];
  const float* Wv = (const float*)d_in[3];
  float* out = (float*)d_out;

  // Workspace carve-up (bf16): kb/qb [B,T,H], vT [B,H,T], packed weights.
  const size_t nQKV = (size_t)B * T * H;            // 1M elements each
  char* ws = (char*)d_ws;
  bf16_t* kb    = (bf16_t*)(ws);
  bf16_t* qb    = (bf16_t*)(ws + nQKV * 2);
  bf16_t* vTb   = (bf16_t*)(ws + nQKV * 4);
  bf16_t* Wpack = (bf16_t*)(ws + nQKV * 6);         // (C/32)*3*4*512 bf16

  const float scale = sqrtf((float)C);              // reference multiplies by C**0.5

  {
    const int total = (C / 32) * 3 * 4 * 32;
    pack_w_kernel<<<(total + 255) / 256, 256, 0, stream>>>(Wk, Wq, Wv, Wpack, C, H);
  }
  {
    const int waves = B * (T / 16);                 // 1024 wave-tiles
    proj_kernel<<<waves / 8, 256, 0, stream>>>(x, Wpack, kb, qb, vTb, B, T, C, H);
  }
  {
    const int waves = B * (T / 16);
    attn_kernel<<<waves / 8, 256, 0, stream>>>(qb, kb, vTb, out, B, T, H, scale);
  }
  (void)in_sizes; (void)n_in; (void)out_size; (void)ws_size;
}
